// ClockworkRNN_89627377533526
// MI455X (gfx1250) — matrix operand order repaired and checked by reading, not compiled, not run on hardware
//
#include <hip/hip_runtime.h>
#include <stdint.h>

typedef __attribute__((ext_vector_type(16))) __bf16       v16bf;
typedef __attribute__((ext_vector_type(8)))  float        v8f;
typedef __attribute__((ext_vector_type(8)))  unsigned int v8u;

typedef __attribute__((address_space(3))) uint16_t lds_u16;

#define B_    64
#define T_    512
#define IN_   512
#define D_    1024
#define KTILE 64

__device__ __forceinline__ uint16_t f2bf(float f) {
    // round-to-nearest-even f32 -> bf16
    uint32_t u = __builtin_bit_cast(uint32_t, f);
    u += 0x7FFFu + ((u >> 16) & 1u);
    return (uint16_t)(u >> 16);
}

// ---------------------------------------------------------------------------
// One-time conversions (outputs are L2-resident for the whole GEMM):
//   X  (B*T x IN f32, row-major)  -> Xbf bf16 same layout          (33.5 MB)
//   W  (IN x D f32)               -> WT  bf16 transposed [d][k]    (1 MB)
//   cw_i (Ki x 128 f32)           -> cwT bf16 transposed [n][k]    (1.18 MB)
// ---------------------------------------------------------------------------
__global__ void x_convert_kernel(const float* __restrict__ X,
                                 uint16_t* __restrict__ Xbf, int total4) {
    int e = blockIdx.x * blockDim.x + threadIdx.x;
    if (e >= total4) return;
    float4 v = ((const float4*)X)[e];
    uint32_t lo = (uint32_t)f2bf(v.x) | ((uint32_t)f2bf(v.y) << 16);
    uint32_t hi = (uint32_t)f2bf(v.z) | ((uint32_t)f2bf(v.w) << 16);
    ((uint2*)Xbf)[e] = make_uint2(lo, hi);
}

__global__ void wt_convert_kernel(const float* __restrict__ W,
                                  uint16_t* __restrict__ WT) {
    int e = blockIdx.x * blockDim.x + threadIdx.x;   // 524288 total
    int d = e & (D_ - 1);
    int k = e >> 10;
    WT[(size_t)d * IN_ + k] = f2bf(W[(size_t)k * D_ + d]);
}

__global__ void cw_convert_kernel(const float* __restrict__ cw,
                                  uint16_t* __restrict__ dst, int Ki) {
    int e = blockIdx.x * blockDim.x + threadIdx.x;
    if (e >= Ki * 128) return;
    int n = e & 127;
    int k = e >> 7;
    dst[(size_t)n * Ki + k] = f2bf(cw[(size_t)k * 128 + n]);
}

// ---------------------------------------------------------------------------
// Phase 1 GEMM: out[r][d] = sum_k Xbf[r][k]*WT[d][k] + bias[d]
// Double-buffered async global->LDS staging (ASYNCcnt), bf16 WMMA compute.
// WG tile 128x128, 8 waves x (16 rows x 128 cols), K staged in 64-deep tiles.
// ---------------------------------------------------------------------------
__global__ void __launch_bounds__(256)
gemm_xw_kernel(const uint16_t* __restrict__ Xbf, const uint16_t* __restrict__ WT,
               const float* __restrict__ bias, float* __restrict__ out) {
    __shared__ uint16_t lA[2][128 * KTILE];   // [m][k] bf16
    __shared__ uint16_t lB[2][128 * KTILE];   // [n][k] bf16 (pre-transposed W)

    const int tid  = threadIdx.x;
    const int wave = tid >> 5;
    const int lane = tid & 31;
    const int ln   = lane & 15;
    const int hi   = lane >> 4;
    const int rowbase = blockIdx.x * 128;
    const int colbase = blockIdx.y * 128;

    uint32_t ldsA[2], ldsB[2];
    ldsA[0] = (uint32_t)(uintptr_t)(lds_u16*)&lA[0][0];
    ldsA[1] = (uint32_t)(uintptr_t)(lds_u16*)&lA[1][0];
    ldsB[0] = (uint32_t)(uintptr_t)(lds_u16*)&lB[0][0];
    ldsB[1] = (uint32_t)(uintptr_t)(lds_u16*)&lB[1][0];

    // issue one stage of async copies: 8 x b128 per thread (A:4, B:4)
    auto issue = [&](int k0, int buf) {
#pragma unroll
        for (int j = 0; j < 4; ++j) {
            int c = tid + 256 * j;
            int row = c >> 3, sub = (c & 7) * 8;
            uint32_t go = (uint32_t)(((rowbase + row) * IN_ + k0 + sub) * 2);
            uint32_t lo = ldsA[buf] + (uint32_t)((row * KTILE + sub) * 2);
            asm volatile("global_load_async_to_lds_b128 %0, %1, %2"
                         :: "v"(lo), "v"(go), "s"(Xbf) : "memory");
        }
#pragma unroll
        for (int j = 0; j < 4; ++j) {
            int c = tid + 256 * j;
            int row = c >> 3, sub = (c & 7) * 8;
            uint32_t go = (uint32_t)(((colbase + row) * IN_ + k0 + sub) * 2);
            uint32_t lo = ldsB[buf] + (uint32_t)((row * KTILE + sub) * 2);
            asm volatile("global_load_async_to_lds_b128 %0, %1, %2"
                         :: "v"(lo), "v"(go), "s"(WT) : "memory");
        }
    };

    v8f acc[8];
#pragma unroll
    for (int i = 0; i < 8; ++i)
#pragma unroll
        for (int j = 0; j < 8; ++j) acc[i][j] = 0.0f;

    issue(0, 0);
    const int NSTAGE = IN_ / KTILE;   // 8
    for (int s = 0; s < NSTAGE; ++s) {
        const int buf = s & 1;
        if (s + 1 < NSTAGE) {
            issue((s + 1) * KTILE, buf ^ 1);
            asm volatile("s_wait_asynccnt 8" ::: "memory");  // stage s complete
        } else {
            asm volatile("s_wait_asynccnt 0" ::: "memory");
        }
        __syncthreads();

#pragma unroll
        for (int kk = 0; kk < KTILE; kk += 32) {
            v8u au;
            {
                const uint32_t* p =
                    (const uint32_t*)&lA[buf][(wave * 16 + ln) * KTILE + kk];
#pragma unroll
                for (int j = 0; j < 4; ++j) au[j]     = p[j + hi * 4];
#pragma unroll
                for (int j = 0; j < 4; ++j) au[4 + j] = p[8 + j + hi * 4];
            }
            v16bf a = __builtin_bit_cast(v16bf, au);
#pragma unroll
            for (int nt = 0; nt < 8; ++nt) {
                v8u bu;
                const uint32_t* p =
                    (const uint32_t*)&lB[buf][(nt * 16 + ln) * KTILE + kk];
#pragma unroll
                for (int j = 0; j < 4; ++j) bu[j]     = p[j + hi * 4];
#pragma unroll
                for (int j = 0; j < 4; ++j) bu[4 + j] = p[8 + j + hi * 4];
                v16bf bf = __builtin_bit_cast(v16bf, bu);
                acc[nt] = __builtin_amdgcn_wmma_f32_16x16x32_bf16(
                    false, a, false, bf, (short)0, acc[nt], false, false);
            }
        }
        __syncthreads();
    }

    const int m0 = rowbase + wave * 16;
#pragma unroll
    for (int nt = 0; nt < 8; ++nt) {
        int c = colbase + nt * 16 + ln;
        float bb = bias[c];
#pragma unroll
        for (int r = 0; r < 8; ++r)
            out[(size_t)(m0 + r + hi * 8) * D_ + c] = acc[nt][r] + bb;
    }
}

// ---------------------------------------------------------------------------
// Phase 2: sequential clockwork scan. 4 WGs x 16 batch rows (recurrence is
// per-batch-row independent -> zero inter-WG sync). h in LDS (f32 master +
// bf16 shadow as WMMA A). x_t read from d_out, h_t overwrites it in place.
// ---------------------------------------------------------------------------
__global__ void __launch_bounds__(256)
cwrnn_scan_kernel(const uint16_t* __restrict__ cwT, float* __restrict__ out) {
    extern __shared__ char smem[];
    float*    hM   = (float*)smem;                 // 16 x 1024 f32  (64 KB)
    float*    part = hM + 16 * D_;                 // 16 x 1024 f32  (64 KB)
    uint16_t* hbf  = (uint16_t*)(part + 16 * D_);  // 16 x 1024 bf16 (32 KB)

    const int tid   = threadIdx.x;
    const int wave  = tid >> 5;
    const int lane  = tid & 31;
    const int ln    = lane & 15;
    const int hi    = lane >> 4;
    const int bbase = blockIdx.x * 16;

    for (int e = tid; e < 16 * D_; e += 256) { hM[e] = 0.0f; hbf[e] = 0; }
    __syncthreads();

    for (int t = 0; t < T_; ++t) {
        unsigned act = 0, act1 = 0;
        for (int i = 0; i < 8; ++i) {
            if ((t & ((1 << i) - 1)) == 0) act |= 1u << i;
            if (((t + 1) & ((1 << i) - 1)) == 0) act1 |= 1u << i;
        }

        // fill pre-activation: active blocks <- x_t, inactive <- h_prev
        for (int e = tid; e < 16 * D_; e += 256) {
            int row = e >> 10, d = e & 1023;
            float v = ((act >> (d >> 7)) & 1)
                          ? out[((size_t)(bbase + row) * T_ + t) * D_ + d]
                          : hM[e];
            part[e] = v;
        }
        __syncthreads();

        // WMMA: part[block i] += h_prev[:, :Ki] @ cw_i  for active blocks
        int tileId = 0;
        for (int i = 0; i < 8; ++i) {
            if (!((act >> i) & 1)) continue;
            const int Ki = (i + 1) * 128;
            const uint16_t* wT = cwT + (size_t)16384 * (i * (i + 1) / 2);
            for (int nt = 0; nt < 8; ++nt) {
                if ((tileId++ & 7) != wave) continue;   // wave-uniform split
                const int col = i * 128 + nt * 16 + ln;
                v8f c;
#pragma unroll
                for (int r = 0; r < 8; ++r) c[r] = part[(r + hi * 8) * D_ + col];
                const uint16_t* bcol = wT + (size_t)(nt * 16 + ln) * Ki;
                for (int k0 = 0; k0 < Ki; k0 += 32) {
                    v8u au;
                    {
                        const uint32_t* p = (const uint32_t*)&hbf[ln * D_ + k0];
#pragma unroll
                        for (int j = 0; j < 4; ++j) au[j]     = p[j + hi * 4];
#pragma unroll
                        for (int j = 0; j < 4; ++j) au[4 + j] = p[8 + j + hi * 4];
                    }
                    v16bf a = __builtin_bit_cast(v16bf, au);
                    v8u bu;
                    {
                        const uint32_t* p = (const uint32_t*)(bcol + k0);
#pragma unroll
                        for (int j = 0; j < 4; ++j) bu[j]     = p[j + hi * 4];
#pragma unroll
                        for (int j = 0; j < 4; ++j) bu[4 + j] = p[8 + j + hi * 4];
                    }
                    v16bf bf = __builtin_bit_cast(v16bf, bu);
                    c = __builtin_amdgcn_wmma_f32_16x16x32_bf16(
                        false, a, false, bf, (short)0, c, false, false);
                }
#pragma unroll
                for (int r = 0; r < 8; ++r) part[(r + hi * 8) * D_ + col] = c[r];
            }
        }
        __syncthreads();

        // h = tanh(part); refresh bf16 shadow; emit h_t (overwrites x_t slot);
        // prefetch next step's active x columns (global_prefetch_b8).
        for (int e = tid; e < 16 * D_; e += 256) {
            int row = e >> 10, d = e & 1023;
            float v = tanhf(part[e]);
            hM[e]  = v;
            hbf[e] = f2bf(v);
            out[((size_t)(bbase + row) * T_ + t) * D_ + d] = v;
            if (t + 1 < T_ && ((act1 >> (d >> 7)) & 1))
                __builtin_prefetch(
                    &out[((size_t)(bbase + row) * T_ + (t + 1)) * D_ + d], 0, 1);
        }
        __syncthreads();
    }
}

// ---------------------------------------------------------------------------
extern "C" void kernel_launch(void* const* d_in, const int* in_sizes, int n_in,
                              void* d_out, int out_size, void* d_ws, size_t ws_size,
                              hipStream_t stream) {
    const float* X    = (const float*)d_in[0];
    const float* W    = (const float*)d_in[1];
    const float* bias = (const float*)d_in[2];
    float* out = (float*)d_out;

    // workspace layout (bf16):  cwT | WT | Xbf   (~36 MB, 16B-aligned chunks)
    uint16_t* cwT = (uint16_t*)d_ws;            // 589,824
    uint16_t* WT  = cwT + 589824;               // 524,288
    uint16_t* Xbf = WT + 524288;                // 16,777,216

    for (int i = 0; i < 8; ++i) {
        int Ki = (i + 1) * 128;
        int total = Ki * 128;
        cw_convert_kernel<<<(total + 255) / 256, 256, 0, stream>>>(
            (const float*)d_in[3 + i], cwT + (size_t)16384 * (i * (i + 1) / 2), Ki);
    }
    wt_convert_kernel<<<(IN_ * D_) / 256, 256, 0, stream>>>(W, WT);
    {
        int total4 = (B_ * T_ * IN_) / 4;
        x_convert_kernel<<<(total4 + 255) / 256, 256, 0, stream>>>(X, Xbf, total4);
    }

    dim3 ggrid((B_ * T_) / 128, D_ / 128);
    gemm_xw_kernel<<<ggrid, 256, 0, stream>>>(Xbf, WT, bias, out);

    // 4 WGs x 256 threads, 160 KB dynamic LDS (within 320 KB/WGP)
    cwrnn_scan_kernel<<<4, 256, 160 * 1024, stream>>>(cwT, out);
}
